// GraphVAE_56573309223970
// MI455X (gfx1250) — matrix operand order repaired and checked by reading, not compiled, not run on hardware
//
#include <hip/hip_runtime.h>
#include <hip/hip_bf16.h>

// ---------------------------------------------------------------------------
// GraphVAE forward for MI455X (gfx1250, wave32).
//   conv GEMMs        : v_wmma_f32_16x16x32_bf16 (bf16 in, f32 acc)
//   edge scatter-add  : global_atomic_pk_add_bf16 (inline asm, 2 adds/atomic)
//   edge_logits       : broadcast of one 4-vector (decoder input is z for all E)
// ---------------------------------------------------------------------------

typedef __attribute__((ext_vector_type(16))) __bf16 v16bf;
typedef __attribute__((ext_vector_type(8)))  float  v8f;

__device__ __forceinline__ unsigned short f2bf(float f) {
  unsigned u = __float_as_uint(f);
  u += 0x7FFFu + ((u >> 16) & 1u);            // round-to-nearest-even
  return (unsigned short)(u >> 16);
}
__device__ __forceinline__ float bf2f(unsigned short h) {
  return __uint_as_float(((unsigned)h) << 16);
}
__device__ __forceinline__ unsigned relu_pk_bf16(unsigned x) {
  unsigned lo = (x & 0x00008000u) ? 0u : (x & 0x0000FFFFu);
  unsigned hi = (x & 0x80000000u) ? 0u : (x & 0xFFFF0000u);
  return lo | hi;
}
__device__ __forceinline__ void atomic_pk_add_bf16(unsigned* addr, unsigned val) {
  // packed bf16 atomic add, no return -> STOREcnt; kernel end waits implicitly
  asm volatile("global_atomic_pk_add_bf16 %0, %1, off" :: "v"(addr), "v"(val) : "memory");
}

// -------------------------------- zero fill --------------------------------
__global__ void zero_kernel(unsigned* __restrict__ p, long long n) {
  long long i = (long long)blockIdx.x * blockDim.x + threadIdx.x;
  long long stride = (long long)gridDim.x * blockDim.x;
  for (; i < n; i += stride) p[i] = 0u;
}

// ------------------- pack W1/W2 into WMMA B-fragment layout -----------------
// Wfrag[cblk 0..7][kb 0..3][lane 0..31][j 0..15] (bf16):
//   element j of lane L = W[kb*32 + (L>>4)*16 + j][cblk*16 + (L&15)]
// so each lane's 16 B-values are 32 contiguous bytes (two b128 loads).
__global__ void prepw_kernel(const float* __restrict__ W1,
                             const float* __restrict__ W2,
                             unsigned short* __restrict__ wf1,
                             unsigned short* __restrict__ wf2) {
  int gid = blockIdx.x * blockDim.x + threadIdx.x;   // 0..32767
  const float* W = (gid < 16384) ? W1 : W2;
  unsigned short* wf = (gid < 16384) ? wf1 : wf2;
  int tid = gid & 16383;
  int j    = tid & 15;
  int lane = (tid >> 4) & 31;
  int kb   = (tid >> 9) & 3;
  int cblk = (tid >> 11) & 7;
  int k   = kb * 32 + (lane >> 4) * 16 + j;          // fan_in index
  int col = cblk * 16 + (lane & 15);                 // fan_out index
  wf[tid] = f2bf(W[k * 128 + col]);
}

// ------------------------------ degree / norm ------------------------------
__global__ void degree_kernel(const int* __restrict__ rows, float* __restrict__ deg, int E) {
  int e = blockIdx.x * blockDim.x + threadIdx.x;
  if (e < E) atomicAdd(&deg[rows[e]], 1.0f);
}
__global__ void norm_kernel(const int* __restrict__ rows, const int* __restrict__ cols,
                            const float* __restrict__ deg, float* __restrict__ norm, int E) {
  int e = blockIdx.x * blockDim.x + threadIdx.x;
  if (e < E) {
    float dr = deg[rows[e]], dc = deg[cols[e]];
    float a = dr > 0.f ? rsqrtf(dr) : 0.f;
    float b = dc > 0.f ? rsqrtf(dc) : 0.f;
    norm[e] = a * b;
  }
}

// -------------------- GCN GEMM: dst = (src @ W + b) * norm ------------------
// 256 threads = 8 waves; block handles 16 rows x 128 cols; wave w -> cols
// [16w,16w+16). K=128 -> 4 x v_wmma_f32_16x16x32_bf16 per wave.
// Requires N % 16 == 0 (N = 500000 -> 31250 blocks, exact).
__global__ void __launch_bounds__(256) gcn_gemm_kernel(
    const float*          __restrict__ srcF,   // f32 source (conv1) or null
    const unsigned short* __restrict__ srcB,   // bf16 source (conv2) or null
    const unsigned short* __restrict__ wfrag,  // [8][4][32][16] bf16
    const float*          __restrict__ bias,   // [128]
    const float*          __restrict__ norm,   // [N]
    unsigned short*       __restrict__ dst,    // [N,128] bf16
    int useBf16Relu) {
  __shared__ unsigned short As[16 * 128];      // 4KB A tile (bf16)
  const int tid = threadIdx.x;
  const long long r0 = (long long)blockIdx.x * 16;

  // --- stage A tile: 2048 bf16, 8 per thread, 16B vector traffic ---
  if (useBf16Relu) {
    const uint4* s4 = (const uint4*)(srcB + r0 * 128);
    uint4 q = s4[tid];
    q.x = relu_pk_bf16(q.x); q.y = relu_pk_bf16(q.y);
    q.z = relu_pk_bf16(q.z); q.w = relu_pk_bf16(q.w);
    ((uint4*)As)[tid] = q;
  } else {
    const float4* s4 = (const float4*)(srcF + r0 * 128);
    float4 a = s4[tid * 2 + 0];
    float4 b = s4[tid * 2 + 1];
    uint4 p;
    p.x = (unsigned)f2bf(a.x) | ((unsigned)f2bf(a.y) << 16);
    p.y = (unsigned)f2bf(a.z) | ((unsigned)f2bf(a.w) << 16);
    p.z = (unsigned)f2bf(b.x) | ((unsigned)f2bf(b.y) << 16);
    p.w = (unsigned)f2bf(b.z) | ((unsigned)f2bf(b.w) << 16);
    ((uint4*)As)[tid] = p;
  }
  __syncthreads();

  const int lane = tid & 31;
  const int cblk = tid >> 5;        // wave id == 16-col block
  const int h    = lane >> 4;       // lane half
  const int n    = lane & 15;       // col within block / A-row M

  const uint4* as4 = (const uint4*)As;
  const uint4* wf4 = (const uint4*)wfrag;

  v8f acc = {0.f, 0.f, 0.f, 0.f, 0.f, 0.f, 0.f, 0.f};
#pragma unroll
  for (int kb = 0; kb < 4; ++kb) {
    v16bf afrag, bfrag;
    // A 16x32 bf16 fragment: M = lane&15; VGPR0-7: K = kb*32 + h*16 + 0..15
    // (matches B packing)  -> two 16B LDS loads
    ((uint4*)&afrag)[0] = as4[n * 16 + kb * 4 + h * 2];
    ((uint4*)&afrag)[1] = as4[n * 16 + kb * 4 + h * 2 + 1];
    // B 32x16 bf16 fragment: pre-packed contiguous 32B per lane
    int wbase = ((cblk * 4 + kb) * 32 + lane) * 2;
    ((uint4*)&bfrag)[0] = wf4[wbase + 0];
    ((uint4*)&bfrag)[1] = wf4[wbase + 1];
    acc = __builtin_amdgcn_wmma_f32_16x16x32_bf16(
        false, afrag, false, bfrag, (short)0, acc, false, false);
  }

  // --- epilogue: +bias, *norm[row], store bf16 ---
  const int col = cblk * 16 + n;
  const float bb = bias[col];
#pragma unroll
  for (int v = 0; v < 8; ++v) {
    long long r = r0 + h * 8 + v;           // D layout: M = v + (lane>=16)*8
    float val = (acc[v] + bb) * norm[r];
    dst[r * 128 + col] = f2bf(val);
  }
}

// ------------- scatter: agg[row[e]] += lin[col[e]]  (packed bf16) ----------
__global__ void __launch_bounds__(64) scatter_kernel(
    const int* __restrict__ rows, const int* __restrict__ cols,
    const unsigned* __restrict__ lin, unsigned* __restrict__ agg) {
  const int e = blockIdx.x;
  const int j = threadIdx.x;                 // 0..63 packed pairs (128 bf16)
  const int r = rows[e];
  const int c = cols[e];
  unsigned val = lin[c * 64 + j];
  atomic_pk_add_bf16(&agg[r * 64 + j], val);
}

// ----------------- column sums of relu(agg) for mean pooling ---------------
__global__ void __launch_bounds__(128) colsum_kernel(
    const unsigned short* __restrict__ agg, float* __restrict__ gsum, int N) {
  const int c = threadIdx.x;
  float s = 0.f;
  for (long long r = blockIdx.x; r < N; r += gridDim.x) {
    float v = bf2f(agg[r * 128 + c]);
    s += (v > 0.f) ? v : 0.f;
  }
  atomicAdd(&gsum[c], s);
}

// ----------------------------- VAE head (tiny) -----------------------------
__global__ void __launch_bounds__(128) head_kernel(
    const float* __restrict__ gsum, const float* __restrict__ eps,
    const float* __restrict__ Wmu, const float* __restrict__ bmu,
    const float* __restrict__ Wlv, const float* __restrict__ blv,
    const float* __restrict__ Wn,  const float* __restrict__ bn,
    const float* __restrict__ We1, const float* __restrict__ be1,
    const float* __restrict__ We2, const float* __restrict__ be2,
    float* __restrict__ out_node, float* __restrict__ out_mu,
    float* __restrict__ out_lv, float* __restrict__ elog, float invN) {
  __shared__ float g[128], z[32], eh[128];
  const int t = threadIdx.x;
  g[t] = gsum[t] * invN;
  __syncthreads();
  if (t < 32) {
    float mu = bmu[t], lv = blv[t];
    for (int c = 0; c < 128; ++c) { mu += g[c] * Wmu[c * 32 + t];
                                    lv += g[c] * Wlv[c * 32 + t]; }
    out_mu[t] = mu;
    out_lv[t] = lv;
    z[t] = mu + eps[t] * expf(0.5f * lv);
  }
  __syncthreads();
  if (t < 16) {
    float s = bn[t];
    for (int l = 0; l < 32; ++l) s += z[l] * Wn[l * 16 + t];
    out_node[t] = s;
  }
  {
    float s = be1[t];
    for (int j = 0; j < 64; ++j) s += z[j & 31] * We1[j * 128 + t];
    eh[t] = (s > 0.f) ? s : 0.f;
  }
  __syncthreads();
  if (t < 4) {
    float s = be2[t];
    for (int c = 0; c < 128; ++c) s += eh[c] * We2[c * 4 + t];
    elog[t] = s;
  }
}

// ------------------ broadcast the 4 edge logits to all E rows --------------
__global__ void broadcast_kernel(const float* __restrict__ elog,
                                 float4* __restrict__ dst, int E) {
  int e = blockIdx.x * blockDim.x + threadIdx.x;
  float4 v = *(const float4*)elog;
  if (e < E) dst[e] = v;
}

// ---------------------------------------------------------------------------
extern "C" void kernel_launch(void* const* d_in, const int* in_sizes, int n_in,
                              void* d_out, int out_size, void* d_ws, size_t ws_size,
                              hipStream_t stream) {
  const float* x   = (const float*)d_in[0];
  const int* eidx  = (const int*)d_in[1];
  const float* eps = (const float*)d_in[2];
  const float* W1  = (const float*)d_in[3];  const float* b1  = (const float*)d_in[4];
  const float* W2  = (const float*)d_in[5];  const float* b2  = (const float*)d_in[6];
  const float* Wmu = (const float*)d_in[7];  const float* bmu = (const float*)d_in[8];
  const float* Wlv = (const float*)d_in[9];  const float* blv = (const float*)d_in[10];
  const float* Wn  = (const float*)d_in[11]; const float* bn  = (const float*)d_in[12];
  const float* We1 = (const float*)d_in[13]; const float* be1 = (const float*)d_in[14];
  const float* We2 = (const float*)d_in[15]; const float* be2 = (const float*)d_in[16];

  const int N = in_sizes[0] / 128;     // 500000 (divisible by 16)
  const int E = in_sizes[1] / 2;       // 500000
  const int* rows = eidx;
  const int* cols = eidx + E;

  // workspace layout (256B-aligned)
  char* ws = (char*)d_ws;
  auto a256 = [](size_t v) { return (v + 255) & ~(size_t)255; };
  size_t off = 0;
  unsigned short* lin  = (unsigned short*)(ws + off); off = a256(off + (size_t)N * 128 * 2);
  unsigned short* agg  = (unsigned short*)(ws + off); off = a256(off + (size_t)N * 128 * 2);
  float* deg           = (float*)(ws + off);          off = a256(off + (size_t)N * 4);
  float* nrm           = (float*)(ws + off);          off = a256(off + (size_t)N * 4);
  float* gsum          = (float*)(ws + off);          off = a256(off + 128 * 4);
  float* elog          = (float*)(ws + off);          off = a256(off + 4 * 4);
  unsigned short* wf1  = (unsigned short*)(ws + off); off = a256(off + 16384 * 2);
  unsigned short* wf2  = (unsigned short*)(ws + off); off = a256(off + 16384 * 2);
  (void)ws_size; (void)n_in; (void)out_size;

  float* fout = (float*)d_out;
  float* out_node = fout;                                  // [16]
  float* out_edge = fout + 16;                             // [E,4]
  float* out_mu   = fout + 16 + (size_t)E * 4;             // [32]
  float* out_lv   = out_mu + 32;                           // [32]

  const int EB = (E + 255) / 256;

  // init
  zero_kernel<<<256, 256, 0, stream>>>((unsigned*)deg, N);
  zero_kernel<<<1, 128, 0, stream>>>((unsigned*)gsum, 128);
  prepw_kernel<<<128, 256, 0, stream>>>(W1, W2, wf1, wf2);

  // degree + symmetric norm (per edge; E == N so norm also scales lin rows)
  degree_kernel<<<EB, 256, 0, stream>>>(rows, deg, E);
  norm_kernel<<<EB, 256, 0, stream>>>(rows, cols, deg, nrm, E);

  // conv1: lin = (x@W1 + b1)*norm ; agg = scatter_add(lin[col] -> row)
  gcn_gemm_kernel<<<N / 16, 256, 0, stream>>>(x, nullptr, wf1, b1, nrm, lin, 0);
  zero_kernel<<<4096, 256, 0, stream>>>((unsigned*)agg, (long long)N * 64);
  scatter_kernel<<<E, 64, 0, stream>>>(rows, cols, (const unsigned*)lin, (unsigned*)agg);

  // conv2: lin = (relu(agg)@W2 + b2)*norm ; agg = scatter_add(...)
  gcn_gemm_kernel<<<N / 16, 256, 0, stream>>>(nullptr, agg, wf2, b2, nrm, lin, 1);
  zero_kernel<<<4096, 256, 0, stream>>>((unsigned*)agg, (long long)N * 64);
  scatter_kernel<<<E, 64, 0, stream>>>(rows, cols, (const unsigned*)lin, (unsigned*)agg);

  // pool + head
  colsum_kernel<<<1024, 128, 0, stream>>>(agg, gsum, N);
  head_kernel<<<1, 128, 0, stream>>>(gsum, eps, Wmu, bmu, Wlv, blv, Wn, bn,
                                     We1, be1, We2, be2,
                                     out_node, out_mu, out_lv, elog, 1.0f / (float)N);

  // edge logits: identical 4-vector for every edge
  broadcast_kernel<<<EB, 256, 0, stream>>>(elog, (float4*)out_edge, E);
}
